// CVNeuralOpKernel_57037165691283
// MI455X (gfx1250) — matrix operand order repaired and checked by reading, not compiled, not run on hardware
//
#include <hip/hip_runtime.h>

// ============================================================================
// Complex NNConv (graph conv with per-edge MLP-generated [16x16] transforms)
// for MI455X / gfx1250.
//
// Roofline: edge MLPs dominate (~85 GFLOP bf16); per-edge W matrices (2x330MB
// if materialized) are kept entirely in VGPRs/LDS by fusing the MLP with the
// einsum+scatter. All weights live in the 320KB/WGP LDS, pre-swizzled into
// WMMA B-operand layout. Matrix math uses v_wmma_f32_16x16x32_bf16 (fp32
// accumulate). Scatter-mean via global f32 atomics into L2-resident buffers.
// ============================================================================

typedef __attribute__((ext_vector_type(16))) __bf16       v16bf;
typedef __attribute__((ext_vector_type(8)))  float        v8f;
typedef __attribute__((ext_vector_type(4)))  unsigned int u32x4;

#define N_NODES 10000
#define N_EDGES 320000
#define CH      16
#define KWID    128
#define EFEAT   6
#define NTILES  (N_EDGES / 16)

// ---- packed bf16 weight layout (in d_ws, then copied to LDS) --------------
// B tile = 32(K) x 16(N) bf16 = 1024 B, stored lane-major: lane L owns 32 B
//   = 8 u32; u32 r holds B[kbase+2r][n] | B[kbase+2r+1][n]<<16,
//   n = L%16, kbase = (L/16)*16.   Tile index within a layer = nb*4 + kb.
#define TILE_B    1024
#define W2_OFF    0
#define W3_OFF    32768
#define W4_OFF    65536
#define MLP_PACK  131072
#define PACK_BYTES 262144          // both MLPs

// ---- workspace layout (bytes) ---------------------------------------------
#define WS_MSGR   262144           // [N_NODES*16] f32
#define WS_MSGI   902144           // [N_NODES*16] f32
#define WS_CNT    1542144          // [N_NODES]    f32
// total 1582144 B (~1.51 MB)

// ---- LDS layout (bytes) ----------------------------------------------------
#define LDS_PACK    0              // 262144: packed bf16 w2/w3/w4, both MLPs
#define LDS_W1      262144         // 2 * 6*128 f32
#define LDS_BIAS    268288         // 2 * (128+128+128+256) f32
#define LDS_WAVE    273408         // per-wave staging
#define ACT_STRIDE  272            // bf16 act row: 128*2=256 + 16B pad (banks)
#define WAVE_STRIDE 6976           // act 4352 | ea 512 | xgr 1024 | xgi 1024 | dst 64
#define NWAVES      6
#define NTHREADS    192
#define LDS_BYTES   (LDS_WAVE + NWAVES * WAVE_STRIDE)   // 315264 <= 320K

struct MlpPtrs {
  const float *w1, *b1, *b2, *b3, *b4, *a1, *a2, *a3;
};

__device__ inline unsigned short f2bf(float f) {
  unsigned int u = __builtin_bit_cast(unsigned int, f);
  u += 0x7fffu + ((u >> 16) & 1u);          // round-to-nearest-even
  return (unsigned short)(u >> 16);
}

__device__ inline v8f splat8(float x) {
  v8f v;
#pragma unroll
  for (int i = 0; i < 8; ++i) v[i] = x;
  return v;
}

__device__ inline void wave_sync() {
  // cross-lane LDS RAW within a wave: LDS pipe is in-order per wave (ISA
  // "LDS: in order"); this just stops compiler reordering.
  asm volatile("" ::: "memory");
  __builtin_amdgcn_wave_barrier();
  asm volatile("" ::: "memory");
}

// A operand, 16x32 bf16 from row-major bf16 activations in LDS.
// Lane L: M = L%16; K runs [16h..16h+15], h = L/16.
__device__ inline v16bf load_A(const char* act, int lane, int kb) {
  const char* p = act + (lane & 15) * ACT_STRIDE + kb * 64 + ((lane >> 4) << 5);
  union { u32x4 q[2]; v16bf v; } U;
  U.q[0] = *(const u32x4*)(p);
  U.q[1] = *(const u32x4*)(p + 16);
  return U.v;
}

// B operand from pre-swizzled packed tile: lane L reads its 32 contiguous B.
__device__ inline v16bf load_B(const char* tileBase, int lane) {
  const char* p = tileBase + lane * 32;
  union { u32x4 q[2]; v16bf v; } U;
  U.q[0] = *(const u32x4*)(p);
  U.q[1] = *(const u32x4*)(p + 16);
  return U.v;
}

// ---------------------------------------------------------------------------
__global__ void zero_ws_kernel(float* __restrict__ p, int n) {
  int stride = gridDim.x * blockDim.x;
  for (int i = blockIdx.x * blockDim.x + threadIdx.x; i < n; i += stride)
    p[i] = 0.0f;
}

// fp32 -> bf16 + swizzle into WMMA-B lane layout for w2/w3/w4 of both MLPs.
__global__ void pack_weights_kernel(const float* __restrict__ w2r,
                                    const float* __restrict__ w3r,
                                    const float* __restrict__ w4r,
                                    const float* __restrict__ w2i,
                                    const float* __restrict__ w3i,
                                    const float* __restrict__ w4i,
                                    unsigned int* __restrict__ out) {
  int stride = gridDim.x * blockDim.x;
  for (int i = blockIdx.x * blockDim.x + threadIdx.x; i < 65536; i += stride) {
    int mlp = i >> 15;
    int j   = i & 32767;
    const float* W; int N; int jj;
    if (j < 8192)       { W = mlp ? w2i : w2r; N = 128; jj = j; }
    else if (j < 16384) { W = mlp ? w3i : w3r; N = 128; jj = j - 8192; }
    else                { W = mlp ? w4i : w4r; N = 256; jj = j - 16384; }
    int tile = jj >> 8, q = jj & 255, lane = q >> 3, r = q & 7;
    int nb = tile >> 2, kb = tile & 3;
    int n = nb * 16 + (lane & 15);
    int k = kb * 32 + ((lane >> 4) << 4) + 2 * r;
    unsigned int lo = f2bf(W[k * N + n]);
    unsigned int hi = f2bf(W[(k + 1) * N + n]);
    out[i] = lo | (hi << 16);
  }
}

// ---------------------------------------------------------------------------
// Fused: per-16-edge tile, run both edge MLPs with WMMA and stream layer-4
// output tiles straight into the message accumulation, then scatter-add.
__global__ __launch_bounds__(NTHREADS, 1)
void edge_mlp_agg_kernel(const float* __restrict__ edge_attr,
                         const int*   __restrict__ edge_index,
                         const float* __restrict__ xr,
                         const float* __restrict__ xi,
                         const char*  __restrict__ packedW,
                         float* __restrict__ msgR,
                         float* __restrict__ msgI,
                         float* __restrict__ cntA,
                         MlpPtrs m0, MlpPtrs m1) {
  extern __shared__ char smem[];
  const int tid  = threadIdx.x;
  const int lane = tid & 31;
  const int wid  = tid >> 5;

  // ---- stage weights / biases into LDS (whole workgroup) ----
  {
    u32x4*       dst = (u32x4*)(smem + LDS_PACK);
    const u32x4* src = (const u32x4*)packedW;
    for (int i = tid; i < PACK_BYTES / 16; i += NTHREADS) dst[i] = src[i];

    float* w1l = (float*)(smem + LDS_W1);
    for (int i = tid; i < EFEAT * KWID; i += NTHREADS) {
      w1l[i]       = m0.w1[i];
      w1l[768 + i] = m1.w1[i];
    }
    float* bl = (float*)(smem + LDS_BIAS);
    for (int i = tid; i < 128; i += NTHREADS) {
      bl[i]        = m0.b1[i]; bl[128 + i] = m0.b2[i]; bl[256 + i] = m0.b3[i];
      bl[640 + i]  = m1.b1[i]; bl[768 + i] = m1.b2[i]; bl[896 + i] = m1.b3[i];
    }
    for (int i = tid; i < 256; i += NTHREADS) {
      bl[384 + i]  = m0.b4[i];
      bl[1024 + i] = m1.b4[i];
    }
  }
  __syncthreads();

  const float A1[2] = { *m0.a1, *m1.a1 };
  const float A2[2] = { *m0.a2, *m1.a2 };
  const float A3[2] = { *m0.a3, *m1.a3 };

  char*  wbase = smem + LDS_WAVE + wid * WAVE_STRIDE;
  char*  act   = wbase;                       // bf16 [16][128] (+pad)
  float* ea    = (float*)(wbase + 4352);      // [16][8] edge features
  float* xgr   = (float*)(wbase + 4864);      // [16][16] gathered xr rows
  float* xgi   = (float*)(wbase + 5888);      // [16][16] gathered xi rows
  int*   dstl  = (int*)(wbase + 6912);        // [16] dst node ids

  const int gwave  = blockIdx.x * NWAVES + wid;
  const int stride = gridDim.x * NWAVES;
  const int col    = lane & 15;
  const int half8  = (lane >> 4) << 3;        // 0 or 8: row offset of D regs

  for (int tile = gwave; tile < NTILES; tile += stride) {
    const int e0 = tile * 16;

    // ---- gather: edge features, src rows, dst ids ----
    if (lane < 16) dstl[lane] = edge_index[N_EDGES + e0 + lane];
    for (int i = lane; i < 96; i += 32) {
      int el = i / 6, k = i - el * 6;
      ea[el * 8 + k] = edge_attr[(e0 + el) * 6 + k];
    }
    for (int i = lane; i < 256; i += 32) {
      int el = i >> 4, c2 = i & 15;
      int s = edge_index[e0 + el];
      xgr[i] = xr[s * 16 + c2];
      xgi[i] = xi[s * 16 + c2];
    }
    wave_sync();

    float msgr[8], msgi[8];
#pragma unroll
    for (int r = 0; r < 8; ++r) { msgr[r] = 0.0f; msgi[r] = 0.0f; }

#pragma unroll
    for (int mlp = 0; mlp < 2; ++mlp) {
      const float* w1l = (const float*)(smem + LDS_W1) + mlp * 768;
      const float* bl  = (const float*)(smem + LDS_BIAS) + mlp * 640;
      const char*  pk  = smem + LDS_PACK + mlp * MLP_PACK;
      const float aa1 = A1[mlp], aa2 = A2[mlp], aa3 = A3[mlp];

      // ---- layer 1 (K=6, VALU) -> bf16 act[16][128] ----
      for (int i = lane; i < 1024; i += 32) {
        int el = i >> 6, n = (i & 63) * 2;
        float s0 = bl[n], s1 = bl[n + 1];
#pragma unroll
        for (int k = 0; k < 6; ++k) {
          float a = ea[el * 8 + k];
          s0 += a * w1l[k * 128 + n];
          s1 += a * w1l[k * 128 + n + 1];
        }
        s0 = s0 >= 0.0f ? s0 : aa1 * s0;
        s1 = s1 >= 0.0f ? s1 : aa1 * s1;
        unsigned int pv = (unsigned int)f2bf(s0) | ((unsigned int)f2bf(s1) << 16);
        *(unsigned int*)(act + el * ACT_STRIDE + n * 2) = pv;
      }
      wave_sync();

      // ---- layers 2 & 3 (128x128, WMMA bf16, PReLU) ----
#pragma unroll
      for (int layer = 0; layer < 2; ++layer) {
        const char*  wp = pk + (layer ? W3_OFF : W2_OFF);
        const float* bb = bl + 128 + layer * 128;
        const float  aa = layer ? aa3 : aa2;

        v16bf A[4];
#pragma unroll
        for (int kb = 0; kb < 4; ++kb) A[kb] = load_A(act, lane, kb);

#pragma unroll
        for (int nb = 0; nb < 8; ++nb) {
          v8f c = splat8(bb[nb * 16 + col]);
#pragma unroll
          for (int kb = 0; kb < 4; ++kb) {
            v16bf B = load_B(wp + (nb * 4 + kb) * TILE_B, lane);
            c = __builtin_amdgcn_wmma_f32_16x16x32_bf16(
                    false, A[kb], false, B, (short)0, c, false, false);
          }
#pragma unroll
          for (int r = 0; r < 8; ++r) {
            float v = c[r];
            v = v >= 0.0f ? v : aa * v;
            int m = r + half8;
            *(unsigned short*)(act + m * ACT_STRIDE + (nb * 16 + col) * 2) =
                f2bf(v);
          }
        }
        wave_sync();
      }

      // ---- layer 4 (128x256) streamed into message accumulation ----
      // N-tile t of layer4 == row t of every W_e: msg[e,o] += x[src][t]*W[t,o]
      v16bf A4[4];
#pragma unroll
      for (int kb = 0; kb < 4; ++kb) A4[kb] = load_A(act, lane, kb);
      const char*  wp4 = pk + W4_OFF;
      const float* b4  = bl + 384;

#pragma unroll
      for (int t = 0; t < 16; ++t) {
        v8f c = splat8(b4[t * 16 + col]);
#pragma unroll
        for (int kb = 0; kb < 4; ++kb) {
          v16bf B = load_B(wp4 + (t * 4 + kb) * TILE_B, lane);
          c = __builtin_amdgcn_wmma_f32_16x16x32_bf16(
                  false, A4[kb], false, B, (short)0, c, false, false);
        }
#pragma unroll
        for (int r = 0; r < 8; ++r) {
          int el = r + half8;                       // edge within tile
          float xrv = xgr[el * 16 + t];
          float xiv = xgi[el * 16 + t];
          if (mlp == 0) {                           // Wr: rr & ri paths
            msgr[r] += xrv * c[r];
            msgi[r] += xiv * c[r];
          } else {                                  // Wi: -ii & ir paths
            msgr[r] -= xiv * c[r];
            msgi[r] += xrv * c[r];
          }
        }
      }
    } // mlp

    // ---- scatter-add combined messages (D layout: e=r+8h, o=col) ----
#pragma unroll
    for (int r = 0; r < 8; ++r) {
      int el = r + half8;
      int d  = dstl[el];
      atomicAdd(msgR + d * 16 + col, msgr[r]);
      atomicAdd(msgI + d * 16 + col, msgi[r]);
    }
    if (lane < 16) atomicAdd(cntA + dstl[lane], 1.0f);
    wave_sync();
  }
}

// ---------------------------------------------------------------------------
__global__ void finalize_kernel(const float* __restrict__ xr,
                                const float* __restrict__ xi,
                                const float* __restrict__ rootR,
                                const float* __restrict__ biasR,
                                const float* __restrict__ rootI,
                                const float* __restrict__ biasI,
                                const float* __restrict__ preluR,
                                const float* __restrict__ preluI,
                                const float* __restrict__ msgR,
                                const float* __restrict__ msgI,
                                const float* __restrict__ cntA,
                                float* __restrict__ out) {
  int id = blockIdx.x * blockDim.x + threadIdx.x;
  if (id >= N_NODES * CH) return;
  int n = id >> 4, c = id & 15;
  float cv = cntA[n];
  cv = cv > 1.0f ? cv : 1.0f;
  float ar = msgR[id] / cv + biasR[c] - biasI[c];
  float ai = msgI[id] / cv + biasR[c] + biasI[c];
#pragma unroll
  for (int k = 0; k < 16; ++k) {
    float xrv = xr[n * 16 + k], xiv = xi[n * 16 + k];
    float rr  = rootR[k * 16 + c], ri = rootI[k * 16 + c];
    ar += xrv * rr - xiv * ri;        // (rr - ii) root part
    ai += xiv * rr + xrv * ri;        // (ri + ir) root part
  }
  float pr = *preluR, pi = *preluI;
  out[id]                = ar >= 0.0f ? ar : pr * ar;
  out[N_NODES * CH + id] = ai >= 0.0f ? ai : pi * ai;
}

// ---------------------------------------------------------------------------
extern "C" void kernel_launch(void* const* d_in, const int* in_sizes, int n_in,
                              void* d_out, int out_size, void* d_ws,
                              size_t ws_size, hipStream_t stream) {
  (void)in_sizes; (void)n_in; (void)out_size; (void)ws_size;

  const float* xr = (const float*)d_in[0];
  const float* xi = (const float*)d_in[1];
  const int*   ei = (const int*)d_in[2];
  const float* ea = (const float*)d_in[3];
  // real params: 4..16, imag params: 17..29 (w1,b1,a1,w2,b2,a2,w3,b3,a3,w4,b4,root,bias)
  const float* rw1 = (const float*)d_in[4];
  const float* rb1 = (const float*)d_in[5];
  const float* ra1 = (const float*)d_in[6];
  const float* rw2 = (const float*)d_in[7];
  const float* rb2 = (const float*)d_in[8];
  const float* ra2 = (const float*)d_in[9];
  const float* rw3 = (const float*)d_in[10];
  const float* rb3 = (const float*)d_in[11];
  const float* ra3 = (const float*)d_in[12];
  const float* rw4 = (const float*)d_in[13];
  const float* rb4 = (const float*)d_in[14];
  const float* rroot = (const float*)d_in[15];
  const float* rbias = (const float*)d_in[16];
  const float* iw1 = (const float*)d_in[17];
  const float* ib1 = (const float*)d_in[18];
  const float* ia1 = (const float*)d_in[19];
  const float* iw2 = (const float*)d_in[20];
  const float* ib2 = (const float*)d_in[21];
  const float* ia2 = (const float*)d_in[22];
  const float* iw3 = (const float*)d_in[23];
  const float* ib3 = (const float*)d_in[24];
  const float* ia3 = (const float*)d_in[25];
  const float* iw4 = (const float*)d_in[26];
  const float* ib4 = (const float*)d_in[27];
  const float* iroot = (const float*)d_in[28];
  const float* ibias = (const float*)d_in[29];
  const float* preluR = (const float*)d_in[30];
  const float* preluI = (const float*)d_in[31];

  char*  ws     = (char*)d_ws;
  unsigned int* packed = (unsigned int*)(ws);
  float* msgR = (float*)(ws + WS_MSGR);
  float* msgI = (float*)(ws + WS_MSGI);
  float* cntA = (float*)(ws + WS_CNT);

  // 1) zero scatter accumulators (msgR|msgI|cnt contiguous = 330000 floats)
  zero_ws_kernel<<<1290, 256, 0, stream>>>(msgR, 330000);

  // 2) pack/convert weights to swizzled bf16
  pack_weights_kernel<<<256, 256, 0, stream>>>(rw2, rw3, rw4, iw2, iw3, iw4,
                                               packed);

  // 3) fused edge MLP + einsum + scatter
  MlpPtrs m0{rw1, rb1, rb2, rb3, rb4, ra1, ra2, ra3};
  MlpPtrs m1{iw1, ib1, ib2, ib3, ib4, ia1, ia2, ia3};
  edge_mlp_agg_kernel<<<400, NTHREADS, LDS_BYTES, stream>>>(
      ea, ei, xr, xi, (const char*)ws, msgR, msgI, cntA, m0, m1);

  // 4) mean + root terms + bias + PReLU -> out_r | out_i
  finalize_kernel<<<(N_NODES * CH + 255) / 256, 256, 0, stream>>>(
      xr, xi, rroot, rbias, iroot, ibias, preluR, preluI, msgR, msgI, cntA,
      (float*)d_out);
}
